// Classifier_64587718197982
// MI455X (gfx1250) — matrix operand order repaired and checked by reading, not compiled, not run on hardware
//
#include <hip/hip_runtime.h>
#include <hip/hip_bf16.h>

typedef _Float16 half_t;
typedef __attribute__((ext_vector_type(16))) _Float16 v16h;
typedef __attribute__((ext_vector_type(8)))  _Float16 v8h;
typedef __attribute__((ext_vector_type(8)))  float    v8f;

// Problem constants
#define TT   8
#define BB   32
#define CC   128
#define HH   32
#define WW   32
#define TB   (TT*BB)          // 256
#define HW   (HH*WW)          // 1024
#define NPIX ((size_t)TB*HW)  // 262144 pixels
#define NELEM (NPIX*CC)       // 33,554,432 activations

#define IN_TILE_HALVES  (4 * 34 * CC)   // 17408 halves = 34816 B (rows h0-1..h0+2, halo w)
#define WT_TILE_HALVES  (CC * CC)       // 16384 halves = 32768 B (one tap: [co][ci])
#define CONV_LDS_BYTES  ((IN_TILE_HALVES + WT_TILE_HALVES) * 2)

union AFrag { v16h h; unsigned int u[8]; v8h h8[2]; };

// gfx1250 async global->LDS copy (GV mode), 16B per lane. Tracked by ASYNCcnt.
__device__ __forceinline__ void async_ld_b128(unsigned lds_off, const void* gaddr) {
  asm volatile("global_load_async_to_lds_b128 %0, %1, off"
               :: "v"(lds_off), "v"((unsigned long long)gaddr) : "memory");
}
__device__ __forceinline__ void wait_asynccnt0() {
  asm volatile("s_wait_asynccnt 0x0" ::: "memory");
}
__device__ __forceinline__ unsigned lds_off_of(const void* p) {
  // generic LDS pointer: high 32 bits = shared aperture, low 32 bits = LDS offset
  return (unsigned)(unsigned long long)p;
}

// ---------------------------------------------------------------------------
// NCHW f32 input -> NHWC f16 activations
// ---------------------------------------------------------------------------
__global__ __launch_bounds__(256) void convert_in_kernel(
    const float* __restrict__ x, half_t* __restrict__ act) {
  size_t g = (size_t)blockIdx.x * 256 + threadIdx.x;   // over NELEM
  if (g >= NELEM) return;
  int c  = (int)(g & (CC - 1));
  size_t p = g >> 7;                 // tb*1024 + hw
  int tb = (int)(p >> 10);
  int hw = (int)(p & (HW - 1));
  act[g] = (half_t)x[((size_t)tb * CC + c) * HW + hw];
}

// ---------------------------------------------------------------------------
// conv weight f32 [O][I][3][3] -> f16 [tap][co][ci]
// ---------------------------------------------------------------------------
__global__ __launch_bounds__(256) void convert_wt_kernel(
    const float* __restrict__ w, half_t* __restrict__ wt) {
  int g = blockIdx.x * 256 + threadIdx.x;  // 9*128*128 = 147456
  if (g >= 9 * CC * CC) return;
  int ci  = g & (CC - 1);
  int co  = (g >> 7) & (CC - 1);
  int tap = g >> 14;
  wt[g] = (half_t)w[((size_t)co * CC + ci) * 9 + tap];
}

// ---------------------------------------------------------------------------
// Implicit-GEMM 3x3 conv, WMMA f16 -> f32.
// act: NHWC f16 [TB][H][W][C]; wt: f16 [9][Co][Ci]; out: NHWC f32 [TB][H][W][Co]
// One block per (tb, h-pair): output tile 64 (2 rows x 32 w) x 128 (co).
// 8 waves: wave = mb(4 M-tiles) + nh(2 N-slabs of 64 cols) -> 4 accumulators/wave.
// Input halo + per-tap weight tile staged via async global->LDS (ASYNCcnt).
// ---------------------------------------------------------------------------
__global__ __launch_bounds__(256) void conv_wmma_kernel(
    const half_t* __restrict__ act,
    const half_t* __restrict__ wt,
    float* __restrict__ out) {
  extern __shared__ half_t smem[];
  half_t* in_lds = smem;                       // [4 rows][34 w][128 ci]
  half_t* wt_lds = smem + IN_TILE_HALVES;      // [128 co][128 ci] for current tap

  const int tid = threadIdx.x;
  const int blk = blockIdx.x;      // 0..4095
  const int tb  = blk >> 4;
  const int h0  = (blk & 15) * 2;

  __builtin_prefetch(wt, 0, 1);    // gfx1250 global_prefetch_b8

  // ---- stage input halo tile (rows h0-1..h0+2, w -1..32, all ci) ----
  for (int e8 = tid; e8 < 4 * 34 * 16; e8 += 256) {
    int ci8 = (e8 & 15) << 3;
    int wh  = (e8 >> 4) % 34;
    int r   = (e8 >> 4) / 34;
    int hh  = h0 + r - 1;
    int ww  = wh - 1;
    half_t* dst = in_lds + (size_t)e8 * 8;
    if (hh >= 0 && hh < HH && ww >= 0 && ww < WW) {
      async_ld_b128(lds_off_of(dst),
                    act + (((size_t)tb * HH + hh) * WW + ww) * CC + ci8);
    } else {
      v8h z = {};
      *reinterpret_cast<v8h*>(dst) = z;        // ds_store zero-fill for halo
    }
  }

  const int lane = tid & 31;
  const int wave = tid >> 5;
  const int mb   = wave & 3;       // M 16-tile (0..3) over 64 rows
  const int nh   = wave >> 2;      // N 64-column slab (0..1)
  const int lg   = lane >> 4;      // lane group (0/1)
  const int ll   = lane & 15;

  const int arow  = mb * 16 + ll;  // A-matrix row within 64-row tile
  const int hsub  = arow >> 5;     // which of the 2 output image rows
  const int wr    = arow & 31;     // image w position

  const v8h* in_v = reinterpret_cast<const v8h*>(in_lds);
  const v8h* wt_v = reinterpret_cast<const v8h*>(wt_lds);

  v8f acc[4] = {{}, {}, {}, {}};

  for (int tap = 0; tap < 9; ++tap) {
    if (tap) __syncthreads();      // previous tap's weight tile fully consumed
    // ---- stage this tap's weight tile [128co][128ci] (coalesced async) ----
    for (int e8 = tid; e8 < WT_TILE_HALVES / 8; e8 += 256) {
      async_ld_b128(lds_off_of(wt_lds + (size_t)e8 * 8),
                    wt + (size_t)tap * CC * CC + (size_t)e8 * 8);
    }
    wait_asynccnt0();              // also covers the input-halo asyncs on tap 0
    __syncthreads();

    const int dh = tap / 3, dw = tap % 3;
    const int rowbase = ((hsub + dh) * 34 + wr + dw) * CC;  // halfword idx in in_lds
#pragma unroll
    for (int cb = 0; cb < 4; ++cb) {
      const int cbase = cb * 32;

      // A fragment (16-bit A 16x32 layout): two contiguous 16B runs per lane:
      //   halves [lg*8 .. lg*8+7] and [16+lg*8 .. 16+lg*8+7] of this K-chunk.
      AFrag a;
      a.h8[0] = in_v[(rowbase + cbase + lg * 8) >> 3];
      a.h8[1] = in_v[(rowbase + cbase + 16 + lg * 8) >> 3];

      // B fragments (16-bit B 32x16 layout): halves [lg*8 .. lg*8+7] and
      // [16+lg*8 .. 16+lg*8+7] of column n -> two contiguous 16B runs per lane.
      AFrag b[4];
#pragma unroll
      for (int j = 0; j < 4; ++j) {
        const int n = nh * 64 + j * 16 + ll;
        const int hbase = n * CC + cbase + lg * 8;
        b[j].h8[0] = wt_v[hbase >> 3];
        b[j].h8[1] = wt_v[(hbase + 16) >> 3];
      }
#pragma unroll
      for (int j = 0; j < 4; ++j) {
        acc[j] = __builtin_amdgcn_wmma_f32_16x16x32_f16(
            false, a.h, false, b[j].h, (short)0, acc[j], false, false);
      }
    }
  }

  // ---- epilogue: C/D layout lanes0-15 M=r, lanes16-31 M=8+r; N=lane%16 ----
  float* obase = out + ((size_t)tb * HH) * WW * CC;
#pragma unroll
  for (int j = 0; j < 4; ++j) {
#pragma unroll
    for (int r = 0; r < 8; ++r) {
      int m   = mb * 16 + lg * 8 + r;        // 0..63
      int hs  = m >> 5;
      int wo  = m & 31;
      obase[(((size_t)(h0 + hs)) * WW + wo) * CC + nh * 64 + j * 16 + ll] = acc[j][r];
    }
  }
}

// ---------------------------------------------------------------------------
// per-channel sum / sumsq over all pixels (TDBN batch stats)
// ---------------------------------------------------------------------------
__global__ __launch_bounds__(256) void zero_kernel(float* p, int n) {
  int i = blockIdx.x * 256 + threadIdx.x;
  if (i < n) p[i] = 0.f;
}

__global__ __launch_bounds__(256) void bn_stats_kernel(
    const float* __restrict__ y, float* __restrict__ sums) {
  __shared__ float red[512];
  int tid = threadIdx.x;
  int c   = tid & (CC - 1);
  int hf  = tid >> 7;
  float s = 0.f, ss = 0.f;
  size_t base = (size_t)blockIdx.x * 1024;   // pixels per block (grid = 256)
  for (int i = hf; i < 1024; i += 2) {
    float v = y[(base + i) * CC + c];
    s += v; ss += v * v;
  }
  red[tid] = s; red[256 + tid] = ss;
  __syncthreads();
  if (tid < 128) {
    atomicAdd(&sums[c],       red[tid] + red[tid + 128]);
    atomicAdd(&sums[CC + c],  red[256 + tid] + red[256 + tid + 128]);
  }
}

// ---------------------------------------------------------------------------
// fused TDBN normalize + LIF recurrence over T, emitting f16 spikes (NHWC)
// ---------------------------------------------------------------------------
__global__ __launch_bounds__(256) void bn_lif_kernel(
    const float* __restrict__ y, const float* __restrict__ sums,
    const float* __restrict__ bw, const float* __restrict__ bb,
    half_t* __restrict__ spikes) {
  size_t g = (size_t)blockIdx.x * 256 + threadIdx.x;  // over B*HW*C = 4194304
  int c  = (int)(g & (CC - 1));
  size_t pb = g >> 7;                 // b*1024 + hw
  int b  = (int)(pb >> 10);
  int hw = (int)(pb & (HW - 1));
  const float Ninv = 1.f / (float)NPIX;
  float mean  = sums[c] * Ninv;
  float var   = sums[CC + c] * Ninv - mean * mean;
  float inv   = rsqrtf(var + 1e-5f);
  float scale = bw[c] * inv;
  float shift = bb[c] - mean * scale;
  float mem = 0.f, spk = 0.f;
  for (int t = 0; t < TT; ++t) {
    size_t idx = ((size_t)(t * BB + b) * HW + hw) * CC + c;
    float xn = y[idx] * scale + shift;
    mem = mem * 0.25f * (1.f - spk) + xn;
    spk = (mem > 0.5f) ? 1.f : 0.f;
    spikes[idx] = (half_t)spk;
  }
}

// ---------------------------------------------------------------------------
// pooled mean over T,H,W per (b,c): feat[b][c]
// ---------------------------------------------------------------------------
__global__ __launch_bounds__(256) void feat_kernel(
    const half_t* __restrict__ sp, float* __restrict__ feat) {
  __shared__ float red[256];
  int b   = blockIdx.x;             // 0..31
  int tid = threadIdx.x;
  int c   = tid & (CC - 1);
  int hf  = tid >> 7;
  float s = 0.f;
  for (int i = hf; i < TT * HW; i += 2) {
    int t  = i >> 10;
    int hw = i & (HW - 1);
    s += (float)sp[((size_t)(t * BB + b) * HW + hw) * CC + c];
  }
  red[tid] = s;
  __syncthreads();
  if (tid < 128)
    feat[b * CC + c] = (red[tid] + red[tid + 128]) * (1.f / (float)(TT * HW));
}

// ---------------------------------------------------------------------------
// final FC: [32,128] @ [128,10]^T + bias -> [32,10]
// ---------------------------------------------------------------------------
__global__ __launch_bounds__(320) void fc_kernel(
    const float* __restrict__ feat, const float* __restrict__ fw,
    const float* __restrict__ fb, float* __restrict__ out) {
  int tid = threadIdx.x;
  if (tid >= 320) return;
  int b = tid / 10, n = tid % 10;
  float s = fb[n];
  for (int c = 0; c < CC; ++c) s += feat[b * CC + c] * fw[n * CC + c];
  out[tid] = s;
}

// ---------------------------------------------------------------------------
extern "C" void kernel_launch(void* const* d_in, const int* in_sizes, int n_in,
                              void* d_out, int out_size, void* d_ws, size_t ws_size,
                              hipStream_t stream) {
  const float* inp    = (const float*)d_in[0];
  const float* conv_w[3] = { (const float*)d_in[1], (const float*)d_in[2],
                             (const float*)d_in[3] };
  const float* bn_w[3]   = { (const float*)d_in[4], (const float*)d_in[6],
                             (const float*)d_in[8] };
  const float* bn_b[3]   = { (const float*)d_in[5], (const float*)d_in[7],
                             (const float*)d_in[9] };
  const float* fc_w = (const float*)d_in[10];
  const float* fc_b = (const float*)d_in[11];
  float* out = (float*)d_out;

  char* ws = (char*)d_ws;
  half_t* actA = (half_t*)ws;                          //  67,108,864 B
  half_t* actB = (half_t*)(ws + (size_t)67108864);     //  67,108,864 B
  float*  conv = (float*) (ws + (size_t)134217728);    // 134,217,728 B
  half_t* wtb  = (half_t*)(ws + (size_t)268435456);    //     294,912 B
  float*  sums = (float*) (ws + (size_t)268730368);    //       1,024 B
  float*  feat = (float*) (ws + (size_t)268731392);    //      16,384 B

  // input -> NHWC f16
  convert_in_kernel<<<(int)((NELEM + 255) / 256), 256, 0, stream>>>(inp, actA);

  half_t* a_in  = actA;
  half_t* a_out = actB;
  for (int l = 0; l < 3; ++l) {
    convert_wt_kernel<<<(9 * CC * CC + 255) / 256, 256, 0, stream>>>(conv_w[l], wtb);
    conv_wmma_kernel<<<TB * HH / 2, 256, CONV_LDS_BYTES, stream>>>(a_in, wtb, conv);
    zero_kernel<<<1, 256, 0, stream>>>(sums, 2 * CC);
    bn_stats_kernel<<<256, 256, 0, stream>>>(conv, sums);
    bn_lif_kernel<<<(BB * HW * CC) / 256, 256, 0, stream>>>(conv, sums, bn_w[l],
                                                            bn_b[l], a_out);
    half_t* t = a_in; a_in = a_out; a_out = t;
  }

  feat_kernel<<<BB, 256, 0, stream>>>(a_in, feat);
  fc_kernel<<<1, 320, 0, stream>>>(feat, fc_w, fc_b, out);
}
